// GATv4_36240934044035
// MI455X (gfx1250) — matrix operand order repaired and checked by reading, not compiled, not run on hardware
//
#include <hip/hip_runtime.h>
#include <hip/hip_bf16.h>

// ---------------- problem constants ----------------
#define B_    16
#define NN_   2000
#define N_    32000        // B_*NN_
#define FIN   128
#define NH    4
#define CH    64
#define HC_   256          // NH*CH
#define EDG   384000       // B*NN*DEG
#define ETOT  (EDG + N_)   // + self loops
#define FC1_  512
#define FC2_  128
#define FCIN  12000        // 6*NN

typedef __attribute__((ext_vector_type(16))) __bf16 bf16x16;
typedef __attribute__((ext_vector_type(8)))  float  f32x8;

__device__ __forceinline__ unsigned short f2bf(float f) {
  unsigned u = __float_as_uint(f);
  unsigned r = (u + 0x7FFFu + ((u >> 16) & 1u)) >> 16;   // RNE
  return (unsigned short)r;
}
__device__ __forceinline__ float bf2f(unsigned short b) {
  return __uint_as_float(((unsigned)b) << 16);
}

// ---------------- conversion kernels ----------------
__global__ void k_f32_to_bf16(const float* __restrict__ src,
                              unsigned short* __restrict__ dst, int n) {
  int i = blockIdx.x * blockDim.x + threadIdx.x;
  if (i < n) dst[i] = f2bf(src[i]);
}

// w: [K,Nw] row-major f32  ->  wt: [Nw,K] row-major bf16
__global__ void k_transpose_bf16(const float* __restrict__ w,
                                 unsigned short* __restrict__ wt, int K, int Nw) {
  int i = blockIdx.x * blockDim.x + threadIdx.x;
  if (i < K * Nw) {
    int k = i / Nw, n = i - k * Nw;
    wt[(size_t)n * K + k] = f2bf(w[i]);
  }
}

// ---------------- x0 = mean(x,-1) : one wave per node ----------------
__global__ void k_mean_x(const float* __restrict__ x, float* __restrict__ x0) {
  int wid  = (blockIdx.x * blockDim.x + threadIdx.x) >> 5;
  int lane = threadIdx.x & 31;
  if (wid >= N_) return;
  const float4* p = reinterpret_cast<const float4*>(x + (size_t)wid * FIN);
  float4 v = p[lane];                       // 32 lanes * 4 = 128
  float s = v.x + v.y + v.z + v.w;
  for (int m = 16; m > 0; m >>= 1) s += __shfl_xor(s, m, 32);
  if (lane == 0) x0[wid] = s * (1.0f / FIN);
}

// ---------------- bf16 WMMA GEMM: C[M,N] = A[M,K] * Bt[N,K]^T ----------------
// One wave computes one 16x16 tile.  A/B tile gathers follow the CDNA5
// 16-bit WMMA VGPR layouts (two 16B loads per lane for each operand).
// MODE 0: write f32 C.   MODE 1: Cb = bf16(relu(C + bias[n])).
template <int MODE>
__global__ void k_gemm_bf16(const unsigned short* __restrict__ A,
                            const unsigned short* __restrict__ Bt,
                            float* __restrict__ C,
                            unsigned short* __restrict__ Cb,
                            const float* __restrict__ bias,
                            int M, int Nn, int K) {
  int wid  = (blockIdx.x * blockDim.x + threadIdx.x) >> 5;
  int lane = threadIdx.x & 31;
  int nTiles = Nn >> 4;
  int mt = wid / nTiles;
  int nt = wid - mt * nTiles;
  if (mt >= (M >> 4)) return;               // wave-uniform: EXEC stays all-ones
  int m0 = mt << 4, n0 = nt << 4;
  int half = lane >> 4, l15 = lane & 15;

  f32x8 acc = {0.f, 0.f, 0.f, 0.f, 0.f, 0.f, 0.f, 0.f};
  const unsigned short* arow = A  + (size_t)(m0 + l15) * K + half * 8;
  const unsigned short* brow = Bt + (size_t)(n0 + l15) * K + half * 8;
  for (int k0 = 0; k0 < K; k0 += 32) {
    union { uint4 q[2]; bf16x16 v; } a, b;
    a.q[0] = *reinterpret_cast<const uint4*>(arow + k0);        // K 0..7 / 8..15
    a.q[1] = *reinterpret_cast<const uint4*>(arow + k0 + 16);   // K 16..23 / 24..31
    b.q[0] = *reinterpret_cast<const uint4*>(brow + k0);        // K 0..7 / 8..15
    b.q[1] = *reinterpret_cast<const uint4*>(brow + k0 + 16);   // K 16..23 / 24..31
    acc = __builtin_amdgcn_wmma_f32_16x16x32_bf16(
        false, a.v, false, b.v, (short)0, acc, false, false);
  }
  int col   = n0 + l15;
  int rbase = m0 + half * 8;                 // D: VGPR i -> row rbase+i
  if (MODE == 0) {
    for (int i = 0; i < 8; ++i)
      C[(size_t)(rbase + i) * Nn + col] = acc[i];
  } else {
    float bv = bias[col];
    for (int i = 0; i < 8; ++i) {
      float v = acc[i] + bv;
      v = v > 0.f ? v : 0.f;
      Cb[(size_t)(rbase + i) * Nn + col] = f2bf(v);
    }
  }
}

// ---------------- attention scores: a_s, a_d per (node, head) ----------------
__global__ void k_att_scores(const float* __restrict__ H,
                             const float* __restrict__ att_s,
                             const float* __restrict__ att_d,
                             float* __restrict__ AS, float* __restrict__ AD) {
  int wid  = (blockIdx.x * blockDim.x + threadIdx.x) >> 5;
  int lane = threadIdx.x & 31;
  if (wid >= N_) return;
  const float4* hp = reinterpret_cast<const float4*>(H + (size_t)wid * HC_);
  const float4* sp = reinterpret_cast<const float4*>(att_s);  // [4,64] flat
  const float4* dp = reinterpret_cast<const float4*>(att_d);
  float4 h0 = hp[lane * 2], h1 = hp[lane * 2 + 1];
  float4 s0 = sp[lane * 2], s1 = sp[lane * 2 + 1];
  float4 d0 = dp[lane * 2], d1 = dp[lane * 2 + 1];
  float ss = h0.x*s0.x + h0.y*s0.y + h0.z*s0.z + h0.w*s0.w
           + h1.x*s1.x + h1.y*s1.y + h1.z*s1.z + h1.w*s1.w;
  float sd = h0.x*d0.x + h0.y*d0.y + h0.z*d0.z + h0.w*d0.w
           + h1.x*d1.x + h1.y*d1.y + h1.z*d1.z + h1.w*d1.w;
  for (int m = 1; m < 8; m <<= 1) {          // reduce 8-lane head groups
    ss += __shfl_xor(ss, m, 32);
    sd += __shfl_xor(sd, m, 32);
  }
  if ((lane & 7) == 0) {
    AS[(size_t)wid * NH + (lane >> 3)] = ss;
    AD[(size_t)wid * NH + (lane >> 3)] = sd;
  }
}

// ---- softmax denominator (shift-free: exp(a)/sum exp(a) == softmax) ----
__global__ void k_edge_denom(const int* __restrict__ esrc, const int* __restrict__ edst,
                             const float* __restrict__ AS, const float* __restrict__ AD,
                             float* __restrict__ DEN) {
  int e = blockIdx.x * blockDim.x + threadIdx.x;
  if (e >= ETOT) return;
  int s, d;
  if (e < EDG) { s = esrc[e]; d = edst[e]; } else { s = d = e - EDG; }
  for (int h = 0; h < NH; ++h) {
    float al = AS[s * NH + h] + AD[d * NH + h];
    al = al > 0.f ? al : 0.2f * al;          // leaky_relu(0.2)
    atomicAdd(&DEN[d * NH + h], __expf(al));
  }
}

// ---- weighted aggregation: one wave per edge, 8 channels per lane ----
__global__ void k_edge_agg(const int* __restrict__ esrc, const int* __restrict__ edst,
                           const float* __restrict__ AS, const float* __restrict__ AD,
                           const float* __restrict__ DEN, const float* __restrict__ H,
                           float* __restrict__ ACC) {
  int e    = (blockIdx.x * blockDim.x + threadIdx.x) >> 5;
  int lane = threadIdx.x & 31;
  if (e >= ETOT) return;
  int s, d;
  if (e < EDG) { s = esrc[e]; d = edst[e]; } else { s = d = e - EDG; }
  int h = lane >> 3;
  float al = AS[s * NH + h] + AD[d * NH + h];
  al = al > 0.f ? al : 0.2f * al;
  float a = __expf(al) / (DEN[d * NH + h] + 1e-16f);
  const float* hs = H + (size_t)s * HC_ + lane * 8;
  float* ap = ACC + (size_t)d * HC_ + lane * 8;
  float4 v0 = *reinterpret_cast<const float4*>(hs);
  float4 v1 = *reinterpret_cast<const float4*>(hs + 4);
  atomicAdd(ap + 0, a * v0.x); atomicAdd(ap + 1, a * v0.y);
  atomicAdd(ap + 2, a * v0.z); atomicAdd(ap + 3, a * v0.w);
  atomicAdd(ap + 4, a * v1.x); atomicAdd(ap + 5, a * v1.y);
  atomicAdd(ap + 6, a * v1.z); atomicAdd(ap + 7, a * v1.w);
}

// ---- bias + relu + pooling projection + bf16 copy for next GEMM ----
__global__ void k_gat_post(const float* __restrict__ ACC, const float* __restrict__ bias,
                           const float* __restrict__ pw, const float* __restrict__ pb,
                           unsigned short* __restrict__ Hb, float* __restrict__ xraw) {
  int n    = (blockIdx.x * blockDim.x + threadIdx.x) >> 5;
  int lane = threadIdx.x & 31;
  if (n >= N_) return;
  int v = lane * 8;
  const float* ap = ACC + (size_t)n * HC_ + v;
  float4 a0 = *reinterpret_cast<const float4*>(ap);
  float4 a1 = *reinterpret_cast<const float4*>(ap + 4);
  float r[8] = {a0.x, a0.y, a0.z, a0.w, a1.x, a1.y, a1.z, a1.w};
  union { unsigned short u[8]; uint4 q; } pk;
  float p = 0.f;
  for (int j = 0; j < 8; ++j) {
    float t = r[j] + bias[v + j];
    t = t > 0.f ? t : 0.f;
    pk.u[j] = f2bf(t);
    p += t * pw[v + j];
  }
  *reinterpret_cast<uint4*>(Hb + (size_t)n * HC_ + v) = pk.q;
  for (int m = 16; m > 0; m >>= 1) p += __shfl_xor(p, m, 32);
  if (lane == 0) xraw[n] = p + pb[0];
}

// ---- demographics -> t[:, 0:6000) (bf16) ----
__global__ void k_demo(const float* __restrict__ sex_emb, const float* __restrict__ mut_emb,
                       const float* __restrict__ age_w, const float* __restrict__ age_b,
                       const float* __restrict__ age, const int* __restrict__ sex,
                       const int* __restrict__ mut, unsigned short* __restrict__ tbf) {
  int i = blockIdx.x * blockDim.x + threadIdx.x;
  if (i >= B_ * NN_) return;
  int b = i / NN_, j = i - b * NN_;
  unsigned short* t = tbf + (size_t)b * FCIN;
  t[j]           = f2bf(sex_emb[sex[b] * NN_ + j]);
  t[NN_ + j]     = f2bf(mut_emb[mut[b] * NN_ + j]);
  t[2 * NN_ + j] = f2bf(age[b] * age_w[j] + age_b[j]);
}

// ---- LayerNorm per (tensor k, batch b) row; writes x_k, ms and bf16 t ----
__global__ void k_layernorm(const float* __restrict__ X0R, const float* __restrict__ X1R,
                            const float* __restrict__ X2R,
                            const float* __restrict__ ln_w, const float* __restrict__ ln_b,
                            float* __restrict__ out_x0, float* __restrict__ out_x1,
                            float* __restrict__ out_x2, float* __restrict__ out_ms,
                            unsigned short* __restrict__ tbf) {
  int r = blockIdx.x;                        // 0..47
  int k = r / B_, b = r - k * B_;
  const float* src = (k == 0 ? X0R : (k == 1 ? X1R : X2R)) + (size_t)b * NN_;
  __shared__ float s1[256], s2[256];
  float a = 0.f, q = 0.f;
  for (int i = threadIdx.x; i < NN_; i += 256) { float v = src[i]; a += v; q += v * v; }
  s1[threadIdx.x] = a; s2[threadIdx.x] = q;
  __syncthreads();
  for (int st = 128; st > 0; st >>= 1) {
    if ((int)threadIdx.x < st) { s1[threadIdx.x] += s1[threadIdx.x + st];
                                 s2[threadIdx.x] += s2[threadIdx.x + st]; }
    __syncthreads();
  }
  float mu  = s1[0] * (1.0f / NN_);
  float var = s2[0] * (1.0f / NN_) - mu * mu;
  float rstd = rsqrtf(var + 1e-5f);
  float* oxk = (k == 0 ? out_x0 : (k == 1 ? out_x1 : out_x2));
  for (int i = threadIdx.x; i < NN_; i += 256) {
    float y = (src[i] - mu) * rstd * ln_w[i] + ln_b[i];
    oxk[(size_t)b * NN_ + i]                     = y;
    out_ms[(size_t)b * 3 * NN_ + k * NN_ + i]    = y;
    tbf[(size_t)b * FCIN + 3 * NN_ + k * NN_ + i] = f2bf(y);
  }
}

// ---- final tiny FC: pred[b] = dot(fc2out[b], fw3) + fb3 ----
__global__ void k_fc3(const unsigned short* __restrict__ f2o, const float* __restrict__ fw3,
                      const float* __restrict__ fb3, float* __restrict__ pred) {
  int b = threadIdx.x;
  if (b >= B_) return;
  float s = 0.f;
  for (int j = 0; j < FC2_; ++j) s += bf2f(f2o[b * FC2_ + j]) * fw3[j];
  pred[b] = s + fb3[0];
}

// ================= host side =================
extern "C" void kernel_launch(void* const* d_in, const int* in_sizes, int n_in,
                              void* d_out, int out_size, void* d_ws, size_t ws_size,
                              hipStream_t stream) {
  const float* x        = (const float*)d_in[0];
  const float* W1       = (const float*)d_in[1];
  const float* att_src1 = (const float*)d_in[2];
  const float* att_dst1 = (const float*)d_in[3];
  const float* b1       = (const float*)d_in[4];
  const float* W2       = (const float*)d_in[5];
  const float* att_src2 = (const float*)d_in[6];
  const float* att_dst2 = (const float*)d_in[7];
  const float* b2       = (const float*)d_in[8];
  const float* pw1      = (const float*)d_in[9];
  const float* pb1      = (const float*)d_in[10];
  const float* pw2      = (const float*)d_in[11];
  const float* pb2      = (const float*)d_in[12];
  const float* ln_w     = (const float*)d_in[13];
  const float* ln_b     = (const float*)d_in[14];
  const float* sex_emb  = (const float*)d_in[15];
  const float* mut_emb  = (const float*)d_in[16];
  const float* age_w    = (const float*)d_in[17];
  const float* age_b    = (const float*)d_in[18];
  const float* fw1      = (const float*)d_in[19];
  const float* fb1      = (const float*)d_in[20];
  const float* fw2      = (const float*)d_in[21];
  const float* fb2      = (const float*)d_in[22];
  const float* fw3      = (const float*)d_in[23];
  const float* fb3      = (const float*)d_in[24];
  const float* age      = (const float*)d_in[25];
  const int*   eidx     = (const int*)d_in[26];
  const int*   sex      = (const int*)d_in[27];
  const int*   mut      = (const int*)d_in[28];
  const int* esrc = eidx;
  const int* edst = eidx + EDG;

  // workspace carve (256B aligned)
  char* ws = (char*)d_ws;
  size_t off = 0;
  auto alloc = [&](size_t bytes) -> char* {
    char* p = ws + off;
    off = (off + bytes + 255) & ~(size_t)255;
    return p;
  };
  unsigned short* XBF  = (unsigned short*)alloc((size_t)N_ * HC_ * 2);  // x bf16, later h_post bf16
  unsigned short* W1T  = (unsigned short*)alloc((size_t)HC_ * FIN * 2);
  unsigned short* W2T  = (unsigned short*)alloc((size_t)HC_ * HC_ * 2);
  unsigned short* FW1T = (unsigned short*)alloc((size_t)FC1_ * FCIN * 2);
  unsigned short* FW2T = (unsigned short*)alloc((size_t)FC2_ * FC1_ * 2);
  float* H    = (float*)alloc((size_t)N_ * HC_ * 4);
  float* ACC  = (float*)alloc((size_t)N_ * HC_ * 4);
  float* AS   = (float*)alloc((size_t)N_ * NH * 4);
  float* AD   = (float*)alloc((size_t)N_ * NH * 4);
  float* DEN  = (float*)alloc((size_t)N_ * NH * 4);
  float* X0R  = (float*)alloc((size_t)N_ * 4);
  float* X1R  = (float*)alloc((size_t)N_ * 4);
  float* X2R  = (float*)alloc((size_t)N_ * 4);
  unsigned short* TBF = (unsigned short*)alloc((size_t)B_ * FCIN * 2);
  unsigned short* F1O = (unsigned short*)alloc((size_t)B_ * FC1_ * 2);
  unsigned short* F2O = (unsigned short*)alloc((size_t)B_ * FC2_ * 2);

  float* out  = (float*)d_out;
  float* o_pred = out;          // 16
  float* o_x0   = out + 16;     // 32000
  float* o_x1   = out + 32016;  // 32000
  float* o_x2   = out + 64016;  // 32000
  float* o_ms   = out + 96016;  // 96000

  const int T = 256;
  auto gemm_blocks = [](int M, int Nn) {
    int waves = (M / 16) * (Nn / 16);
    return (waves * 32 + 255) / 256;
  };

  // ---- conversions ----
  k_f32_to_bf16<<<(N_ * FIN + T - 1) / T, T, 0, stream>>>(x, XBF, N_ * FIN);
  k_transpose_bf16<<<(FIN * HC_ + T - 1) / T, T, 0, stream>>>(W1, W1T, FIN, HC_);
  k_transpose_bf16<<<(HC_ * HC_ + T - 1) / T, T, 0, stream>>>(W2, W2T, HC_, HC_);
  k_transpose_bf16<<<(FCIN * FC1_ + T - 1) / T, T, 0, stream>>>(fw1, FW1T, FCIN, FC1_);
  k_transpose_bf16<<<(FC1_ * FC2_ + T - 1) / T, T, 0, stream>>>(fw2, FW2T, FC1_, FC2_);

  // ---- x0 = mean(x,-1) ----
  k_mean_x<<<N_ / 8, T, 0, stream>>>(x, X0R);

  // ---- GAT layer 1 ----
  k_gemm_bf16<0><<<gemm_blocks(N_, HC_), T, 0, stream>>>(XBF, W1T, H, nullptr, nullptr, N_, HC_, FIN);
  k_att_scores<<<N_ / 8, T, 0, stream>>>(H, att_src1, att_dst1, AS, AD);
  hipMemsetAsync(DEN, 0, (size_t)N_ * NH * 4, stream);
  hipMemsetAsync(ACC, 0, (size_t)N_ * HC_ * 4, stream);
  k_edge_denom<<<(ETOT + T - 1) / T, T, 0, stream>>>(esrc, edst, AS, AD, DEN);
  k_edge_agg<<<ETOT / 8, T, 0, stream>>>(esrc, edst, AS, AD, DEN, H, ACC);
  k_gat_post<<<N_ / 8, T, 0, stream>>>(ACC, b1, pw1, pb1, XBF, X1R);  // XBF := h1 bf16

  // ---- GAT layer 2 ----
  k_gemm_bf16<0><<<gemm_blocks(N_, HC_), T, 0, stream>>>(XBF, W2T, H, nullptr, nullptr, N_, HC_, HC_);
  k_att_scores<<<N_ / 8, T, 0, stream>>>(H, att_src2, att_dst2, AS, AD);
  hipMemsetAsync(DEN, 0, (size_t)N_ * NH * 4, stream);
  hipMemsetAsync(ACC, 0, (size_t)N_ * HC_ * 4, stream);
  k_edge_denom<<<(ETOT + T - 1) / T, T, 0, stream>>>(esrc, edst, AS, AD, DEN);
  k_edge_agg<<<ETOT / 8, T, 0, stream>>>(esrc, edst, AS, AD, DEN, H, ACC);
  k_gat_post<<<N_ / 8, T, 0, stream>>>(ACC, b2, pw2, pb2, XBF, X2R);

  // ---- head: demo features, layernorms (fills t), MLP ----
  k_demo<<<(B_ * NN_ + T - 1) / T, T, 0, stream>>>(sex_emb, mut_emb, age_w, age_b,
                                                   age, sex, mut, TBF);
  k_layernorm<<<48, T, 0, stream>>>(X0R, X1R, X2R, ln_w, ln_b,
                                    o_x0, o_x1, o_x2, o_ms, TBF);
  k_gemm_bf16<1><<<gemm_blocks(B_, FC1_), T, 0, stream>>>(TBF, FW1T, nullptr, F1O, fb1, B_, FC1_, FCIN);
  k_gemm_bf16<1><<<gemm_blocks(B_, FC2_), T, 0, stream>>>(F1O, FW2T, nullptr, F2O, fb2, B_, FC2_, FC1_);
  k_fc3<<<1, 32, 0, stream>>>(F2O, fw3, fb3, o_pred);

  (void)in_sizes; (void)n_in; (void)out_size; (void)ws_size;
}
